// MHSAX_59983513256290
// MI455X (gfx1250) — matrix operand order repaired and checked by reading, not compiled, not run on hardware
//
#include <hip/hip_runtime.h>

// ---------- types ----------
typedef __bf16 bf16;
typedef __attribute__((ext_vector_type(16))) __bf16 v16bf;
typedef __attribute__((ext_vector_type(8)))  float  v8f;
typedef __attribute__((ext_vector_type(4)))  unsigned int uintx4;

union Frag {            // 32-byte WMMA operand fragment (8 VGPRs of packed bf16)
    uintx4 u[2];
    v16bf  v;
};

#define BATCH 16
#define CH    512
#define NHEAD 8
#define DH    64
#define WD    32
#define HD    32
#define NPOS  1024      // W*H
#define DEXT  128       // 2*DH : [q;pos] / [k;q] packed contraction dim

static __device__ __forceinline__ v8f wmma_bf16(v16bf a, v16bf b, v8f c) {
    // D = A(16x32 bf16) x B(32x16 bf16) + C(16x16 f32)
    return __builtin_amdgcn_wmma_f32_16x16x32_bf16(false, a, false, b,
                                                   (short)0, c, false, false);
}

// async 16B global -> LDS copy (ASYNCcnt-tracked, no VGPR data path)
static __device__ __forceinline__ void async_cp16(const bf16* gbase,
                                                  unsigned goff_bytes,
                                                  void* lds_ptr) {
    unsigned lds = (unsigned)(size_t)lds_ptr;
    unsigned long long sbase = (unsigned long long)(size_t)gbase;
    asm volatile("global_load_async_to_lds_b128 %0, %1, %2"
                 :: "v"(lds), "v"(goff_bytes), "s"(sbase)
                 : "memory");
}
static __device__ __forceinline__ void wait_async0() {
    asm volatile("s_wait_asynccnt 0x0" ::: "memory");
}

// =====================================================================
// Kernel 1: k (depthwise 3x3, replicate pad), v, pos -> packed bf16.
//   Bext[bh][n][0:64]    = k
//   Aext[bh][n][64:128]  = pos (rel_h + rel_w)
//   Vn  [bh][d][n]       = v              (n-contiguous for PV A-fragments)
// =====================================================================
__global__ __launch_bounds__(256) void prep_kernel(
        const float* __restrict__ x,  const float* __restrict__ wk,
        const float* __restrict__ bk, const float* __restrict__ wv,
        const float* __restrict__ bv, const float* __restrict__ rel_h,
        const float* __restrict__ rel_w,
        bf16* __restrict__ Aext, bf16* __restrict__ Bext, bf16* __restrict__ Vn)
{
    __shared__ float xs[64][192];            // 64 channels x 6 w-rows x 32 y (48KB)
    const int bh = blockIdx.y, b = bh >> 3, h = bh & 7;
    const int n0 = blockIdx.x * 128, w0 = n0 >> 5;   // 4 w-rows per tile
    const int tid = threadIdx.x;

    const float* xb = x + (size_t)(b * CH + h * DH) * NPOS;
    for (int idx = tid; idx < 64 * 192; idx += 256) {
        int c = idx / 192, rem = idx - c * 192;
        int lw = rem >> 5, y = rem & 31;
        int g = w0 - 1 + lw; g = g < 0 ? 0 : (g > 31 ? 31 : g);   // replicate pad (w)
        xs[c][rem] = xb[c * NPOS + g * 32 + y];
    }
    __syncthreads();

    // k-conv + pos : thread owns fixed d = tid&63, sweeps n
    const int d = tid & 63;
    float wk9[9];
#pragma unroll
    for (int t = 0; t < 9; ++t) wk9[t] = wk[(h * DH + d) * 9 + t];
    const float bkd = bk[h * DH + d];
    const float* rh = rel_h + (h * DH + d) * HD;
    const float* rw = rel_w + (h * DH + d) * WD;
    bf16* Arow = Aext + (size_t)bh * NPOS * DEXT;
    bf16* Brow = Bext + (size_t)bh * NPOS * DEXT;

    for (int idx = tid; idx < 8192; idx += 256) {
        int nr = idx >> 6, wr = nr >> 5, y = nr & 31;
        float acc = bkd;
#pragma unroll
        for (int i = 0; i < 3; ++i)
#pragma unroll
            for (int j = 0; j < 3; ++j) {
                int yy = y - 1 + j; yy = yy < 0 ? 0 : (yy > 31 ? 31 : yy); // pad (y)
                acc += xs[d][(wr + i) * 32 + yy] * wk9[i * 3 + j];
            }
        int n = n0 + nr;
        Brow[(size_t)n * DEXT + d]      = (bf16)acc;
        Arow[(size_t)n * DEXT + 64 + d] = (bf16)(rh[y] + rw[w0 + wr]);
    }

    // v = x*wv + bv, stored [d][n]
    for (int idx = tid; idx < 8192; idx += 256) {
        int dv = idx >> 7, nr = idx & 127, wr = nr >> 5, y = nr & 31;
        float vv = xs[dv][(wr + 1) * 32 + y] * wv[h * DH + dv] + bv[h * DH + dv];
        Vn[(size_t)bh * DH * NPOS + (size_t)dv * NPOS + n0 + nr] = (bf16)vv;
    }
}

// =====================================================================
// Kernel 2: x [b][c][n] f32  ->  Xbt [b][n][c] bf16 (LDS tile transpose)
// =====================================================================
__global__ __launch_bounds__(256) void xpose_kernel(
        const float* __restrict__ x, bf16* __restrict__ Xbt)
{
    __shared__ float t[64][65];
    const int b = blockIdx.z, c0 = blockIdx.y * 64, n0 = blockIdx.x * 64;
    const int tid = threadIdx.x;
#pragma unroll
    for (int k = 0; k < 16; ++k) {
        int idx = tid + k * 256, cl = idx >> 6, nl = idx & 63;
        t[cl][nl] = x[((size_t)b * CH + c0 + cl) * NPOS + n0 + nl];
    }
    __syncthreads();
#pragma unroll
    for (int k = 0; k < 16; ++k) {
        int idx = tid + k * 256, nl = idx >> 6, cl = idx & 63;
        Xbt[((size_t)b * NPOS + n0 + nl) * CH + c0 + cl] = (bf16)t[cl][nl];
    }
}

// Kernel 3: wq f32 -> bf16
__global__ __launch_bounds__(256) void cvtw_kernel(
        const float* __restrict__ wq, bf16* __restrict__ Wb)
{
    int idx = blockIdx.x * 256 + threadIdx.x;
    if (idx < CH * CH) Wb[idx] = (bf16)wq[idx];
}

// =====================================================================
// Kernel 4: Q = wq @ x + bq  (WMMA bf16 GEMM, M=o=512, N=n=1024/b, K=c=512)
//   writes q into Aext[..][n][0:64] and Bext[..][n][64:128]
// =====================================================================
__global__ __launch_bounds__(128) void qgemm_kernel(
        const bf16* __restrict__ Wb, const bf16* __restrict__ Xbt,
        const float* __restrict__ bq,
        bf16* __restrict__ Aext, bf16* __restrict__ Bext)
{
    const int lane = threadIdx.x & 31, wave = threadIdx.x >> 5;
    const int b = blockIdx.z, o0 = blockIdx.y * 64 + wave * 16, n0 = blockIdx.x * 64;
    const int l16 = lane & 15;
    const bool lolane = lane < 16;
    const int selA = lolane ? 0 : 8;     // A-matrix K select per lane half
    const int selB = lolane ? 0 : 8;     // B-matrix K select per lane half

    v8f acc[4] = {};
    const bf16* wrow = Wb + (size_t)(o0 + l16) * CH;
    const bf16* xb   = Xbt + (size_t)b * NPOS * CH;

    for (int k0 = 0; k0 < CH; k0 += 32) {
        Frag a;
        const uintx4* ap = (const uintx4*)(wrow + k0 + selA);
        a.u[0] = ap[0]; a.u[1] = ap[2];                  // K and K+16 chunks
#pragma unroll
        for (int ns = 0; ns < 4; ++ns) {
            Frag bb;
            const uintx4* bp =
                (const uintx4*)(xb + (size_t)(n0 + ns * 16 + l16) * CH + k0 + selB);
            bb.u[0] = bp[0]; bb.u[1] = bp[2];
            acc[ns] = wmma_bf16(a.v, bb.v, acc[ns]);
        }
    }

    const int hb = lolane ? 0 : 8;
    float bias[8];
#pragma unroll
    for (int r = 0; r < 8; ++r) bias[r] = bq[o0 + r + hb];
    const int h = o0 >> 6, d0 = (o0 & 63) + hb;
#pragma unroll
    for (int ns = 0; ns < 4; ++ns) {
        int n = n0 + ns * 16 + l16;
        union { uintx4 q; bf16 e[8]; } st;
#pragma unroll
        for (int r = 0; r < 8; ++r) st.e[r] = (bf16)(acc[ns][r] + bias[r]);
        size_t base = ((size_t)(b * NHEAD + h) * NPOS + n) * DEXT;
        *(uintx4*)(Aext + base + d0)      = st.q;   // q rows of [q;pos]
        *(uintx4*)(Bext + base + 64 + d0) = st.q;   // q rows of [k;q]
    }
}

// =====================================================================
// Kernel 5: fused flash attention per (b,h).
//   E^T[j,i] = sum_{dd<128} Bext[j][dd]*Aext[i][dd]  (cc+cp in ONE GEMM, K=128)
//   online softmax over j (per-lane, lane = column i), then
//   out^T[d,i] += sum_j Vn[d][j] * P^T[j,i]
//   K/Q-stream + V tiles staged in LDS with double-buffered ASYNC DMA
//   (global_load_async_to_lds_b128) so all 4 waves share one copy.
// =====================================================================
#define BT_PITCH (DEXT + 8)   // 272B rows -> conflict-free ds_load_b128
#define VT_PITCH 40           // 80B rows  -> conflict-free ds_load_b128

__global__ __launch_bounds__(128) void attn_kernel(
        const bf16* __restrict__ Aext, const bf16* __restrict__ Bext,
        const bf16* __restrict__ Vn, float* __restrict__ out)
{
    __shared__ bf16 Bt[2][32][BT_PITCH];   // [buf][j-row][dext]  (~17 KB)
    __shared__ bf16 Vt[2][64][VT_PITCH];   // [buf][d-row][j]     (~10 KB)

    const int tid = threadIdx.x;
    const int lane = tid & 31, wave = tid >> 5;
    const int bh = blockIdx.y, i0 = blockIdx.x * 64 + wave * 16;
    const int l16 = lane & 15;
    const bool lolane = lane < 16;
    const int selA = lolane ? 0 : 8, selB = lolane ? 0 : 8, selV = lolane ? 0 : 16;

    const bf16* Ae = Aext + (size_t)bh * NPOS * DEXT;
    const bf16* Be = Bext + (size_t)bh * NPOS * DEXT;
    const bf16* Vp = Vn + (size_t)bh * DH * NPOS;

    // stage one 32-column tile (jb) into LDS buffer nb: 512+256 x 16B, split
    // across the 128 threads -> 6 async-DMA instructions per wave per tile
    auto stage = [&](int jb, int nb) {
#pragma unroll
        for (int r = 0; r < 4; ++r) {          // Bt: 32 rows x 256B
            int idx = tid + r * 128;
            int row = idx >> 4, c16 = idx & 15;
            unsigned goff = (unsigned)(((jb + row) * DEXT + c16 * 8) * 2);
            async_cp16(Be, goff, &Bt[nb][row][c16 * 8]);
        }
#pragma unroll
        for (int r = 0; r < 2; ++r) {          // Vt: 64 rows x 64B
            int idx = tid + r * 128;
            int row = idx >> 2, c = idx & 3;
            unsigned goff = (unsigned)((row * NPOS + jb + c * 8) * 2);
            async_cp16(Vp, goff, &Vt[nb][row][c * 8]);
        }
    };

    // persistent B-operand fragments: Aext columns i0..i0+15, K = 0..127
    Frag bfr[4];
#pragma unroll
    for (int ks = 0; ks < 4; ++ks) {
        const uintx4* p = (const uintx4*)(Ae + (size_t)(i0 + l16) * DEXT + ks * 32 + selB);
        bfr[ks].u[0] = p[0]; bfr[ks].u[1] = p[2];
    }

    stage(0, 0);                               // prologue DMA for tile 0

    v8f oacc[4] = {};
    float m = -1e30f, l = 0.f;

    for (int jt = 0; jt < 32; ++jt) {
        const int jb = jt * 32, cur = jt & 1;

        wait_async0();        // our DMA for buf[cur] done
        __syncthreads();      // everyone's DMA done + prev compute on buf[cur^1] done
        if (jt + 1 < 32) stage(jb + 32, cur ^ 1);   // overlap next tile's DMA

        // --- energy (transposed): two 16x16 subtiles, K=128 each ---
        v8f es0 = {}, es1 = {};
        const bf16* ar0 = &Bt[cur][l16][selA];
        const bf16* ar1 = &Bt[cur][16 + l16][selA];
#pragma unroll
        for (int ks = 0; ks < 4; ++ks) {
            Frag a0, a1;
            const uintx4* q0 = (const uintx4*)(ar0 + ks * 32);
            a0.u[0] = q0[0]; a0.u[1] = q0[2];
            const uintx4* q1 = (const uintx4*)(ar1 + ks * 32);
            a1.u[0] = q1[0]; a1.u[1] = q1[2];
            es0 = wmma_bf16(a0.v, bfr[ks].v, es0);
            es1 = wmma_bf16(a1.v, bfr[ks].v, es1);
        }

        // --- online softmax over j (per-lane scalars; pair-lane combine) ---
        float pm = es0[0];
#pragma unroll
        for (int r = 1; r < 8; ++r) pm = fmaxf(pm, es0[r]);
#pragma unroll
        for (int r = 0; r < 8; ++r) pm = fmaxf(pm, es1[r]);
        float tmax = fmaxf(pm, __shfl_xor(pm, 16, 32));
        float mnew = fmaxf(m, tmax);
        float corr = __expf(m - mnew);
        float p0a[8], p1a[8], psum = 0.f;
#pragma unroll
        for (int r = 0; r < 8; ++r) {
            p0a[r] = __expf(es0[r] - mnew);
            p1a[r] = __expf(es1[r] - mnew);
            psum += p0a[r] + p1a[r];
        }
        float tsum = psum + __shfl_xor(psum, 16, 32);
        l = l * corr + tsum; m = mnew;
#pragma unroll
        for (int t = 0; t < 4; ++t) oacc[t] *= corr;

        // --- build P^T B-fragment (K=32 j-values, N=16 i) via half-swap ---
        float sh[8];
#pragma unroll
        for (int u = 0; u < 8; ++u) {
            float tmp = lolane ? p1a[u] : p0a[u];
            sh[u] = __shfl_xor(tmp, 16, 32);
        }
        Frag pf;
#pragma unroll
        for (int e = 0; e < 8; ++e) pf.v[e]     = (bf16)(lolane ? p0a[e] : sh[e]);
#pragma unroll
        for (int e = 0; e < 8; ++e) pf.v[8 + e] = (bf16)(lolane ? sh[e] : p1a[e]);

        // --- PV: out^T[d,i] accumulate, A = V rows d from LDS ---
#pragma unroll
        for (int t = 0; t < 4; ++t) {
            Frag vf;
            const uintx4* vp = (const uintx4*)(&Vt[cur][t * 16 + l16][selV]);
            vf.u[0] = vp[0]; vf.u[1] = vp[1];
            oacc[t] = wmma_bf16(vf.v, pf.v, oacc[t]);
        }
    }

    // --- epilogue: divide by row sum, coalesced stores (lane -> n) ---
    const float inv = 1.f / l;
    const int b = bh >> 3, h = bh & 7, hb = lolane ? 0 : 8;
#pragma unroll
    for (int t = 0; t < 4; ++t)
#pragma unroll
        for (int r = 0; r < 8; ++r) {
            int d = t * 16 + r + hb;
            out[((size_t)(b * CH + h * DH + d)) * NPOS + i0 + l16] = oacc[t][r] * inv;
        }
}

// =====================================================================
extern "C" void kernel_launch(void* const* d_in, const int* in_sizes, int n_in,
                              void* d_out, int out_size, void* d_ws, size_t ws_size,
                              hipStream_t stream)
{
    const float* x     = (const float*)d_in[0];
    const float* wq    = (const float*)d_in[1];
    const float* bq    = (const float*)d_in[2];
    const float* wk    = (const float*)d_in[3];
    const float* bk    = (const float*)d_in[4];
    const float* wv    = (const float*)d_in[5];
    const float* bv    = (const float*)d_in[6];
    const float* rel_h = (const float*)d_in[7];
    const float* rel_w = (const float*)d_in[8];

    char* ws = (char*)d_ws;
    bf16* Aext = (bf16*)(ws);                       // 128*1024*128*2 = 32 MiB
    bf16* Bext = (bf16*)(ws + (size_t)33554432);    // 32 MiB
    bf16* Vn   = (bf16*)(ws + (size_t)67108864);    // 16 MiB
    bf16* Xbt  = (bf16*)(ws + (size_t)83886080);    // 16 MiB
    bf16* Wb   = (bf16*)(ws + (size_t)100663296);   // 0.5 MiB

    prep_kernel <<<dim3(8, 128),    256, 0, stream>>>(x, wk, bk, wv, bv, rel_h, rel_w,
                                                      Aext, Bext, Vn);
    xpose_kernel<<<dim3(16, 8, 16), 256, 0, stream>>>(x, Xbt);
    cvtw_kernel <<<dim3(1024),      256, 0, stream>>>(wq, Wb);
    qgemm_kernel<<<dim3(16, 8, 16), 128, 0, stream>>>(Wb, Xbt, bq, Aext, Bext);
    attn_kernel <<<dim3(16, 128),   128, 0, stream>>>(Aext, Bext, Vn, (float*)d_out);
}
